// PRGModule_54262616818010
// MI455X (gfx1250) — compile-verified, matrix operand order checked by reading, not run on hardware
//
#include <hip/hip_runtime.h>

typedef __attribute__((ext_vector_type(16))) _Float16 v16h;
typedef __attribute__((ext_vector_type(8)))  float    v8f;
typedef __attribute__((ext_vector_type(4)))  float    v4f;

#define B_DIM   2
#define S_DIM   1024
#define D_DIM   1024
#define C_DIM   32
#define HID_DIM 64
#define H_DIM   16

// ---------------------------------------------------------------------------
// Prep kernel: xc = x@Wc + bc (per row), then
//   A[row,:] = xc @ W1[:C] + b1   (b1 folded here)
//   Bm[row,:] = xc @ W1[C:]
// 2048 blocks x 64 threads; totally negligible vs. the pair kernel.
// ---------------------------------------------------------------------------
__global__ __launch_bounds__(64) void prg_prep_kernel(
    const float* __restrict__ x,  const float* __restrict__ Wc,
    const float* __restrict__ bc, const float* __restrict__ W1,
    const float* __restrict__ b1, float* __restrict__ Aout,
    float* __restrict__ Bout)
{
  __shared__ float xs[D_DIM];
  __shared__ float xcs[C_DIM];
  const int row = blockIdx.x;            // b*S + s
  const int tid = threadIdx.x;
  const float* xrow = x + (size_t)row * D_DIM;
  for (int d = tid; d < D_DIM; d += 64) xs[d] = xrow[d];
  __syncthreads();
  if (tid < C_DIM) {
    float acc = bc[tid];
#pragma unroll 8
    for (int d = 0; d < D_DIM; ++d) acc += xs[d] * Wc[d * C_DIM + tid];
    xcs[tid] = acc;
  }
  __syncthreads();
  float av = b1[tid];
  float bv = 0.0f;
#pragma unroll
  for (int c = 0; c < C_DIM; ++c) {
    const float xc = xcs[c];
    av += xc * W1[c * HID_DIM + tid];
    bv += xc * W1[(C_DIM + c) * HID_DIM + tid];
  }
  Aout[(size_t)row * HID_DIM + tid] = av;
  Bout[(size_t)row * HID_DIM + tid] = bv;
}

// ---------------------------------------------------------------------------
// Pair kernel: out[b,h,i,j] = sum_k gelu(A[b,i,k]+Bm[b,j,k]) * W2[k,h] + b2[h]
// D = W2^T (16x64, A-operand, f16) x h^T (64x16 j's, B-operand, f16), f32 acc,
// via 2x V_WMMA_F32_16X16X32_F16 per 16-j tile. gelu tanh goes through the
// CDNA5 TRANS pipe (v_tanh_f32) so it co-executes with the WMMA pipe.
// ---------------------------------------------------------------------------
__device__ __forceinline__ float fast_tanh(float x) {
#if __has_builtin(__builtin_amdgcn_tanhf)
  return __builtin_amdgcn_tanhf(x);      // single v_tanh_f32 TRANS op
#else
  return tanhf(x);
#endif
}

__device__ __forceinline__ float gelu_tanh(float v) {
  const float v3 = v * v * v;
  const float t  = fast_tanh(0.7978845608028654f * (v + 0.044715f * v3));
  return 0.5f * v * (1.0f + t);
}

__global__ __launch_bounds__(256) void prg_pair_kernel(
    const float* __restrict__ Amat,   // [B*S, 64]  (b1 folded in)
    const float* __restrict__ Bmat,   // [B*S, 64]
    const float* __restrict__ W2,     // [64, 16]
    const float* __restrict__ b2,     // [16]
    float* __restrict__ out)          // [B, H, S, S]
{
  const int row  = blockIdx.x;            // b*S + i
  const int b    = row >> 10;
  const int i    = row & (S_DIM - 1);
  const int tid  = threadIdx.x;
  const int lane = tid & 31;
  const int wv   = tid >> 5;              // wave id within block (0..7)
  const int jn   = lane & 15;             // N index (j within tile) / M row for A
  const int hlf  = lane >> 4;             // wave half selects K sub-pattern
  const int klo  = hlf << 3;              // K pattern base: 0 or 8

  // W2^T as WMMA A-operand: lane holds M=jn (head), 16 f16 K-values in the
  // hardware A layout: e<8 -> k=klo+e, e>=8 -> k=klo+e+8. wa1 covers K=32..63.
  v16h wa0, wa1;
#pragma unroll
  for (int e = 0; e < 16; ++e) {
    const int k = klo + ((e < 8) ? e : (e + 8));
    wa0[e] = (_Float16)W2[k * H_DIM + jn];
    wa1[e] = (_Float16)W2[(k + 32) * H_DIM + jn];
  }

  // Per-lane A-row values (uniform across the block), same K pattern,
  // loaded once and kept in registers for all 8 tiles this wave handles.
  const float* arow = Amat + (size_t)row * HID_DIM;
  float av[32];
#pragma unroll
  for (int q = 0; q < 4; ++q) {
    const v4f lo = *(const v4f*)(arow + klo + q * 16);
    const v4f hi = *(const v4f*)(arow + klo + q * 16 + 4);
#pragma unroll
    for (int t = 0; t < 4; ++t) { av[q * 8 + t] = lo[t]; av[q * 8 + 4 + t] = hi[t]; }
  }

  // Accumulator seed = b2[h]; D layout: VGPR r -> head r (lanes 0-15) / r+8.
  v8f cinit;
#pragma unroll
  for (int r = 0; r < 8; ++r) cinit[r] = b2[hlf * 8 + r];

  const float* bbase = Bmat + ((size_t)b << 10) * HID_DIM;
  float* obase = out + (((size_t)(b * H_DIM + hlf * 8)) << 20) + ((size_t)i << 10);

  for (int jt = wv; jt < (S_DIM / 16); jt += 8) {
    const int j = (jt << 4) + jn;
    const float* bp = bbase + ((size_t)j << 6);
    if (jt + 8 < (S_DIM / 16)) __builtin_prefetch(bp + (128 << 6), 0, 1);

    // 8 x b128 loads of Bm in the B-operand K pattern, fused add + gelu.
    float g[32];
#pragma unroll
    for (int q = 0; q < 4; ++q) {
      const v4f lo = *(const v4f*)(bp + klo + q * 16);
      const v4f hi = *(const v4f*)(bp + klo + q * 16 + 4);
#pragma unroll
      for (int t = 0; t < 4; ++t) {
        g[q * 8 + t]     = gelu_tanh(av[q * 8 + t] + lo[t]);
        g[q * 8 + 4 + t] = gelu_tanh(av[q * 8 + 4 + t] + hi[t]);
      }
    }
    v16h hb0, hb1;
#pragma unroll
    for (int e = 0; e < 16; ++e) {
      hb0[e] = (_Float16)g[e];
      hb1[e] = (_Float16)g[16 + e];
    }

    v8f acc = cinit;
    acc = __builtin_amdgcn_wmma_f32_16x16x32_f16(
        false, wa0, false, hb0, (short)0, acc, false, false);
    acc = __builtin_amdgcn_wmma_f32_16x16x32_f16(
        false, wa1, false, hb1, (short)0, acc, false, false);

    // Store: VGPR r -> out[b, hlf*8+r, i, j]; lanes give 2x64B contiguous runs.
    float* op = obase + j;
#pragma unroll
    for (int r = 0; r < 8; ++r) op[(size_t)r << 20] = acc[r];
  }
}

extern "C" void kernel_launch(void* const* d_in, const int* in_sizes, int n_in,
                              void* d_out, int out_size, void* d_ws, size_t ws_size,
                              hipStream_t stream) {
  (void)in_sizes; (void)n_in; (void)out_size; (void)ws_size;
  const float* x  = (const float*)d_in[0];
  const float* Wc = (const float*)d_in[1];
  const float* bc = (const float*)d_in[2];
  const float* W1 = (const float*)d_in[3];
  const float* b1 = (const float*)d_in[4];
  const float* W2 = (const float*)d_in[5];
  const float* b2 = (const float*)d_in[6];
  float* out = (float*)d_out;

  float* Aws = (float*)d_ws;                                   // 512 KB
  float* Bws = Aws + (size_t)B_DIM * S_DIM * HID_DIM;          // 512 KB

  prg_prep_kernel<<<B_DIM * S_DIM, 64, 0, stream>>>(x, Wc, bc, W1, b1, Aws, Bws);
  prg_pair_kernel<<<B_DIM * S_DIM, 256, 0, stream>>>(Aws, Bws, W2, b2, out);
}
